// PointNeXtSetAbstraction_90108413870612
// MI455X (gfx1250) — matrix operand order repaired and checked by reading, not compiled, not run on hardware
//
#include <hip/hip_runtime.h>

// ---------------- types for CDNA5 WMMA ----------------
typedef __bf16 bf16_t;
typedef __attribute__((ext_vector_type(16))) __bf16 v16bf;
typedef __attribute__((ext_vector_type(8)))  __bf16 v8bf;
typedef __attribute__((ext_vector_type(8)))  float  v8f;

// ---------------- problem constants ----------------
#define NB    4          // batches
#define NPTS  16384      // input points per batch
#define NS    2048       // NPOINT
#define KS    32         // NSAMPLE
#define CIN   67         // 3 + 64
#define CO    128        // channels
#define MROWS (NB*NS*KS) // 262144 GEMM rows
#define R2    0.04f      // RADIUS^2
#define BN_EPS 1e-5f

// =====================================================================
// 1. Farthest point sampling: one WG per batch, coords resident in LDS
//    (192KB — CDNA5 WGP has 320KB LDS). dist[] lives in registers.
// =====================================================================
extern "C" __global__ __launch_bounds__(1024)
void fps_kernel(const float* __restrict__ xyz,   // (B,3,N)
                int*   __restrict__ fps_idx,     // (B,S)
                float* __restrict__ nxyz_bs3,    // (B,S,3)
                float* __restrict__ out_nxyz)    // d_out part 1: (B,3,S)
{
    extern __shared__ float smem[];
    float* sx   = smem;
    float* sy   = sx + NPTS;
    float* sz   = sy + NPTS;
    float* s_rv = sz + NPTS;          // 32 wave maxima
    int*   s_ri = (int*)(s_rv + 32);  // 32 wave arg indices
    int*   s_far= s_ri + 32;          // broadcast slot

    const int b    = blockIdx.x;
    const int tid  = threadIdx.x;
    const int lane = tid & 31;
    const int wave = tid >> 5;
    const float* xb = xyz + (size_t)b * 3 * NPTS;

    for (int n = tid; n < NPTS; n += 1024) {
        sx[n] = xb[n];
        sy[n] = xb[NPTS + n];
        sz[n] = xb[2 * NPTS + n];
    }
    __syncthreads();

    float dist[16];
#pragma unroll
    for (int j = 0; j < 16; ++j) dist[j] = 1e10f;

    int farthest = 0;
    for (int i = 0; i < NS; ++i) {
        // record current sample (scan outputs incoming index, like reference)
        if (tid < 3) {
            float c = (tid == 0) ? sx[farthest] : (tid == 1) ? sy[farthest] : sz[farthest];
            nxyz_bs3[((size_t)b * NS + i) * 3 + tid] = c;
            out_nxyz[(size_t)b * 3 * NS + (size_t)tid * NS + i] = c;
            if (tid == 0) fps_idx[b * NS + i] = farthest;
        }
        const float cx = sx[farthest], cy = sy[farthest], cz = sz[farthest];

        float bestv = -1.0f;
        int   besti = 0;
#pragma unroll
        for (int j = 0; j < 16; ++j) {
            const int n = j * 1024 + tid;
            const float dx = sx[n] - cx, dy = sy[n] - cy, dz = sz[n] - cz;
            const float d2 = dx * dx + dy * dy + dz * dz;
            dist[j] = fminf(dist[j], d2);
            if (dist[j] > bestv) { bestv = dist[j]; besti = n; }
        }
        // wave32 argmax (prefer lower index on ties, like jnp.argmax)
#pragma unroll
        for (int offd = 16; offd > 0; offd >>= 1) {
            const float ov = __shfl_down(bestv, offd);
            const int   oi = __shfl_down(besti, offd);
            if (ov > bestv || (ov == bestv && oi < besti)) { bestv = ov; besti = oi; }
        }
        if (lane == 0) { s_rv[wave] = bestv; s_ri[wave] = besti; }
        __syncthreads();
        if (tid < 32) {
            bestv = s_rv[tid]; besti = s_ri[tid];
#pragma unroll
            for (int offd = 16; offd > 0; offd >>= 1) {
                const float ov = __shfl_down(bestv, offd);
                const int   oi = __shfl_down(besti, offd);
                if (ov > bestv || (ov == bestv && oi < besti)) { bestv = ov; besti = oi; }
            }
            if (tid == 0) s_far[0] = besti;
        }
        __syncthreads();
        farthest = s_far[0];
    }
}

// =====================================================================
// 2. Ball query: one wave per query; ballot+popcount prefix emits the
//    first 32 in-radius indices in ascending order (== sort+slice).
// =====================================================================
extern "C" __global__ __launch_bounds__(256)
void ballquery_kernel(const float* __restrict__ xyz,       // (B,3,N)
                      const float* __restrict__ nxyz_bs3,  // (B,S,3)
                      int* __restrict__ ballidx)           // (B,S,KS)
{
    const int q    = blockIdx.x * 8 + (threadIdx.x >> 5);
    const int lane = threadIdx.x & 31;
    const int b    = q >> 11;
    const float* xb = xyz + (size_t)b * 3 * NPTS;
    const float* nq = nxyz_bs3 + (size_t)q * 3;
    const float qx = nq[0], qy = nq[1], qz = nq[2];
    int* outp = ballidx + (size_t)q * KS;

    int cnt = 0;
    int firstIdx = 0;
    for (int c0 = 0; c0 < NPTS && cnt < KS; c0 += 32) {
        const int n = c0 + lane;
        const float dx = xb[n] - qx;
        const float dy = xb[NPTS + n] - qy;
        const float dz = xb[2 * NPTS + n] - qz;
        const float d2 = dx * dx + dy * dy + dz * dz;
        const unsigned m = __builtin_amdgcn_ballot_w32(d2 <= R2);
        if (cnt == 0 && m) firstIdx = c0 + __builtin_ctz(m);
        if ((m >> lane) & 1u) {
            const int pos = cnt + __popc(m & ((1u << lane) - 1u));
            if (pos < KS) outp[pos] = n;
        }
        cnt += __popc(m);
    }
    if (cnt < KS) {  // pad with first hit (reference's `first` fill)
        if (cnt == 0 && lane == 0) outp[0] = firstIdx;
        const int start = (cnt == 0) ? 1 : cnt;
        for (int p = start + lane; p < KS; p += 32) outp[p] = firstIdx;
    }
}

// =====================================================================
// 3. Weight pre-pack into per-lane WMMA B fragments, contiguous k order (same as A):
//    B[k][n] = w[n][k];  K = kt*32 + (lane>>4)*16 + e,  N = nt*16 + (lane&15)
//    wpack[((kt*8+nt)*32 + lane)*16 + e]
// =====================================================================
extern "C" __global__ void packw_kernel(const float* __restrict__ w, int csrc,
                                        bf16_t* __restrict__ wp)
{
    const int e_lin = blockIdx.x * blockDim.x + threadIdx.x;  // < 16384
    const int e    = e_lin & 15;
    const int lane = (e_lin >> 4) & 31;
    const int tile = e_lin >> 9;
    const int nt   = tile & 7;
    const int kt   = tile >> 3;
    const int K    = kt * 32 + ((lane >> 4) << 4) + e;
    const int Ncol = nt * 16 + (lane & 15);
    const float v  = (K < csrc) ? w[(size_t)Ncol * csrc + K] : 0.0f;
    wp[e_lin] = (bf16_t)v;
}

// =====================================================================
// 4. Gather + concat -> bf16 GEMM input, rows (b,s,k), cols [gxyz|gpts|0pad]
// =====================================================================
extern "C" __global__ __launch_bounds__(256)
void gather_kernel(const float* __restrict__ xyz,       // (B,3,N)
                   const float* __restrict__ points,    // (B,64,N)
                   const float* __restrict__ nxyz_bs3,  // (B,S,3)
                   const int* __restrict__ ballidx,     // (B,S,KS)
                   bf16_t* __restrict__ A0)             // (MROWS,128)
{
    const int r  = blockIdx.x * blockDim.x + threadIdx.x;  // < MROWS
    const int b  = r >> 16;
    const int bs = r >> 5;
    const int idx = ballidx[r];
    const float* xb = xyz + (size_t)b * 3 * NPTS;
    const float cx = nxyz_bs3[bs * 3 + 0];
    const float cy = nxyz_bs3[bs * 3 + 1];
    const float cz = nxyz_bs3[bs * 3 + 2];
    bf16_t* row = A0 + (size_t)r * CO;
    row[0] = (bf16_t)(xb[idx] - cx);
    row[1] = (bf16_t)(xb[NPTS + idx] - cy);
    row[2] = (bf16_t)(xb[2 * NPTS + idx] - cz);
    const float* pb = points + (size_t)b * 64 * NPTS + idx;
#pragma unroll 4
    for (int c = 0; c < 64; ++c) row[3 + c] = (bf16_t)pb[(size_t)c * NPTS];
#pragma unroll
    for (int c = CIN; c < CO; ++c) row[c] = (bf16_t)0.0f;
}

// =====================================================================
// 5. bf16 WMMA GEMM (M=262144, N=128, K=128) fused with per-channel
//    sum / sum^2 reduction for BatchNorm.
//    Packed weights (32KB) are staged global -> LDS ONCE per block with
//    the CDNA5 async-to-LDS path (ASYNCcnt + s_wait_asynccnt), then the
//    inner loop reads B fragments from LDS while A streams from global.
// =====================================================================
extern "C" __global__ __launch_bounds__(256)
void gemm_bn_kernel(const bf16_t* __restrict__ A,   // (MROWS,128) bf16
                    const bf16_t* __restrict__ Wp,  // packed B fragments (16384)
                    float* __restrict__ Y,          // (MROWS,128) f32
                    float* __restrict__ stats)      // [0..127]=sum [128..255]=sumsq
{
    __shared__ __align__(16) bf16_t s_wp[16384];    // 32 KB packed B fragments
    __shared__ float s_sum[CO];
    __shared__ float s_sq[CO];
    const int tid = threadIdx.x;
    if (tid < CO) { s_sum[tid] = 0.f; s_sq[tid] = 0.f; }

    // ---- async copy of packed weights: global -> LDS (ISA 08, ASYNCcnt) ----
    // LDS aperture rule (ISA 10.2): low 32 bits of a generic LDS pointer are
    // the LDS byte offset, which is exactly what the async VDST VGPR expects.
    {
        const unsigned lds_base = (unsigned)(unsigned long long)(uintptr_t)(void*)s_wp;
#pragma unroll
        for (int it = 0; it < 8; ++it) {
            const int chunk = tid + it * 256;                 // 16-byte chunk id
            const unsigned lds_addr = lds_base + chunk * 16;
            const bf16_t* gsrc = Wp + (size_t)chunk * 8;      // 8 bf16 = 16 B
            asm volatile("global_load_async_to_lds_b128 %0, %1, off"
                         :: "v"(lds_addr), "v"(gsrc)
                         : "memory");
        }
        asm volatile("s_wait_asynccnt 0x0" ::: "memory");
    }
    __syncthreads();

    const int lane  = tid & 31;
    const int wave  = tid >> 5;
    const int mtile = blockIdx.x * 8 + wave;
    const int mrow  = lane & 15;      // also the N coordinate of C/D
    const int hi    = lane >> 4;
    const bf16_t* arow = A + (size_t)(mtile * 16 + mrow) * CO;

    const v8f vzero = {0.f, 0.f, 0.f, 0.f, 0.f, 0.f, 0.f, 0.f};
    v8f acc[8];
#pragma unroll
    for (int nt = 0; nt < 8; ++nt) acc[nt] = vzero;

#pragma unroll
    for (int kt = 0; kt < 4; ++kt) {
        // A fragment, contiguous K order (matches packed B):
        // elems 0..15 -> K = kt*32 + hi*16 + e
        const int k0 = kt * 32 + hi * 16;
        const v8bf a0 = *(const v8bf*)(arow + k0);
        const v8bf a1 = *(const v8bf*)(arow + k0 + 8);
        v16bf af;
#pragma unroll
        for (int i2 = 0; i2 < 8; ++i2) { af[i2] = a0[i2]; af[8 + i2] = a1[i2]; }
#pragma unroll
        for (int nt = 0; nt < 8; ++nt) {
            const v16bf bfrag =
                *(const v16bf*)(s_wp + (size_t)(((kt * 8 + nt) * 32 + lane)) * 16);
            acc[nt] = __builtin_amdgcn_wmma_f32_16x16x32_bf16(
                false, af, false, bfrag, (short)0, acc[nt], false, false);
        }
    }

    // epilogue: store Y, accumulate BN statistics
#pragma unroll
    for (int nt = 0; nt < 8; ++nt) {
        const int ch = nt * 16 + mrow;
        float s = 0.f, q = 0.f;
#pragma unroll
        for (int v = 0; v < 8; ++v) {
            const float val = acc[nt][v];
            const int r = mtile * 16 + hi * 8 + v;   // D layout: M = v + 8*hi
            Y[(size_t)r * CO + ch] = val;
            s += val; q += val * val;
        }
        atomicAdd(&s_sum[ch], s);
        atomicAdd(&s_sq[ch], q);
    }
    __syncthreads();
    if (tid < CO) {
        atomicAdd(&stats[tid], s_sum[tid]);
        atomicAdd(&stats[CO + tid], s_sq[tid]);
    }
}

// =====================================================================
// 6. BN helpers
// =====================================================================
extern "C" __global__ void zero_stats_kernel(float* stats)
{
    if (threadIdx.x < 2 * CO) stats[threadIdx.x] = 0.f;
}

extern "C" __global__ void bn_finalize_kernel(float* stats, const float* __restrict__ g,
                                              const float* __restrict__ b, float invcnt)
{
    const int c = threadIdx.x;
    if (c < CO) {
        const float mu  = stats[c] * invcnt;
        const float var = stats[CO + c] * invcnt - mu * mu;
        const float sc  = g[c] * rsqrtf(var + BN_EPS);
        stats[2 * CO + c] = sc;
        stats[3 * CO + c] = b[c] - mu * sc;
    }
}

// mode 0: x = relu(bn(y));            write xres(f32) + outb(bf16)
// mode 1: h = relu(bn(y));            write outb(bf16) only
// mode 2: x = relu(bn(y) + xres);     write xres(f32) + outb(bf16)
extern "C" __global__ __launch_bounds__(256)
void bn_apply_kernel(const float* __restrict__ Y, const float* __restrict__ stats,
                     float* __restrict__ xres, bf16_t* __restrict__ outb, int mode)
{
    const float* sc = stats + 2 * CO;
    const float* bi = stats + 3 * CO;
    const size_t i4 = (size_t)blockIdx.x * blockDim.x + threadIdx.x;
    if (i4 >= (size_t)MROWS * CO / 4) return;
    const size_t i = i4 * 4;
    const int ch = (int)(i & (CO - 1));
    const float4 y = *(const float4*)(Y + i);
    float v0 = y.x * sc[ch + 0] + bi[ch + 0];
    float v1 = y.y * sc[ch + 1] + bi[ch + 1];
    float v2 = y.z * sc[ch + 2] + bi[ch + 2];
    float v3 = y.w * sc[ch + 3] + bi[ch + 3];
    if (mode == 2) {
        const float4 xo = *(const float4*)(xres + i);
        v0 += xo.x; v1 += xo.y; v2 += xo.z; v3 += xo.w;
    }
    v0 = fmaxf(v0, 0.f); v1 = fmaxf(v1, 0.f);
    v2 = fmaxf(v2, 0.f); v3 = fmaxf(v3, 0.f);
    if (mode != 1) {
        float4 o; o.x = v0; o.y = v1; o.z = v2; o.w = v3;
        *(float4*)(xres + i) = o;
    }
    outb[i + 0] = (bf16_t)v0; outb[i + 1] = (bf16_t)v1;
    outb[i + 2] = (bf16_t)v2; outb[i + 3] = (bf16_t)v3;
}

// =====================================================================
// 7. Max over the K=32 samples, transposed store to (B, CO, S)
// =====================================================================
extern "C" __global__ __launch_bounds__(256)
void maxpool_kernel(const float* __restrict__ xres, float* __restrict__ out)
{
    const int t  = blockIdx.x * blockDim.x + threadIdx.x;  // < NB*NS*CO
    const int ch = t & (CO - 1);
    const int bs = t >> 7;
    const float* p = xres + (size_t)bs * KS * CO + ch;
    float m = -3.4e38f;
#pragma unroll 8
    for (int k = 0; k < KS; ++k) m = fmaxf(m, p[(size_t)k * CO]);
    const int b = bs >> 11, s = bs & 2047;
    out[(size_t)b * CO * NS + (size_t)ch * NS + s] = m;
}

// =====================================================================
// host-side launch
// =====================================================================
extern "C" void kernel_launch(void* const* d_in, const int* in_sizes, int n_in,
                              void* d_out, int out_size, void* d_ws, size_t ws_size,
                              hipStream_t stream)
{
    const float* xyz    = (const float*)d_in[0];   // (4,3,16384)
    const float* points = (const float*)d_in[1];   // (4,64,16384)
    const float* w_proj = (const float*)d_in[2];   // (128,67)
    const float* g_proj = (const float*)d_in[3];
    const float* b_proj = (const float*)d_in[4];
    const float* w1     = (const float*)d_in[5];   // (2,128,128)
    const float* w2     = (const float*)d_in[6];   // (2,128,128)
    const float* g1     = (const float*)d_in[7];   // (2,128)
    const float* b1     = (const float*)d_in[8];
    const float* g2     = (const float*)d_in[9];
    const float* b2     = (const float*)d_in[10];
    float* out = (float*)d_out;

    char* ws = (char*)d_ws;
    size_t off = 0;
    auto alloc = [&](size_t bytes) -> char* {
        char* p = ws + off;
        off += (bytes + 255) & ~(size_t)255;
        return p;
    };
    int*    fps_idx = (int*)   alloc((size_t)NB * NS * 4);
    float*  nxyz    = (float*) alloc((size_t)NB * NS * 3 * 4);
    int*    ballidx = (int*)   alloc((size_t)NB * NS * KS * 4);
    bf16_t* wp_proj = (bf16_t*)alloc(16384 * 2);
    bf16_t* wp_11   = (bf16_t*)alloc(16384 * 2);
    bf16_t* wp_21   = (bf16_t*)alloc(16384 * 2);
    bf16_t* wp_12   = (bf16_t*)alloc(16384 * 2);
    bf16_t* wp_22   = (bf16_t*)alloc(16384 * 2);
    float*  stats   = (float*) alloc(4 * CO * 4);
    bf16_t* xb      = (bf16_t*)alloc((size_t)MROWS * CO * 2);
    bf16_t* hb      = (bf16_t*)alloc((size_t)MROWS * CO * 2);
    float*  y       = (float*) alloc((size_t)MROWS * CO * 4);
    float*  xres    = (float*) alloc((size_t)MROWS * CO * 4);

    const size_t fps_smem = (size_t)3 * NPTS * 4 + 32 * 4 + 32 * 4 + 16;
    fps_kernel<<<NB, 1024, fps_smem, stream>>>(xyz, fps_idx, nxyz, out);
    ballquery_kernel<<<(NB * NS) / 8, 256, 0, stream>>>(xyz, nxyz, ballidx);

    packw_kernel<<<64, 256, 0, stream>>>(w_proj,          CIN, wp_proj);
    packw_kernel<<<64, 256, 0, stream>>>(w1,              CO,  wp_11);
    packw_kernel<<<64, 256, 0, stream>>>(w2,              CO,  wp_21);
    packw_kernel<<<64, 256, 0, stream>>>(w1 + CO * CO,    CO,  wp_12);
    packw_kernel<<<64, 256, 0, stream>>>(w2 + CO * CO,    CO,  wp_22);

    gather_kernel<<<MROWS / 256, 256, 0, stream>>>(xyz, points, nxyz, ballidx, xb);

    const float invcnt = 1.0f / (float)MROWS;
    const int gemm_blocks = MROWS / 16 / 8;          // 2048
    const int bn_blocks   = (MROWS * CO / 4) / 256;  // 32768

    // projection: x = bn_act(A0 @ w_proj^T)
    zero_stats_kernel<<<1, 256, 0, stream>>>(stats);
    gemm_bn_kernel<<<gemm_blocks, 256, 0, stream>>>(xb, wp_proj, y, stats);
    bn_finalize_kernel<<<1, 128, 0, stream>>>(stats, g_proj, b_proj, invcnt);
    bn_apply_kernel<<<bn_blocks, 256, 0, stream>>>(y, stats, xres, xb, 0);

    const bf16_t* wp1[2] = {wp_11, wp_12};
    const bf16_t* wp2[2] = {wp_21, wp_22};
    for (int d = 0; d < 2; ++d) {
        zero_stats_kernel<<<1, 256, 0, stream>>>(stats);
        gemm_bn_kernel<<<gemm_blocks, 256, 0, stream>>>(xb, wp1[d], y, stats);
        bn_finalize_kernel<<<1, 128, 0, stream>>>(stats, g1 + d * CO, b1 + d * CO, invcnt);
        bn_apply_kernel<<<bn_blocks, 256, 0, stream>>>(y, stats, xres, hb, 1);

        zero_stats_kernel<<<1, 256, 0, stream>>>(stats);
        gemm_bn_kernel<<<gemm_blocks, 256, 0, stream>>>(hb, wp2[d], y, stats);
        bn_finalize_kernel<<<1, 128, 0, stream>>>(stats, g2 + d * CO, b2 + d * CO, invcnt);
        bn_apply_kernel<<<bn_blocks, 256, 0, stream>>>(y, stats, xres, xb, 2);
    }

    maxpool_kernel<<<(NB * NS * CO) / 256, 256, 0, stream>>>(xres, out + NB * 3 * NS);

    (void)in_sizes; (void)n_in; (void)out_size; (void)ws_size; (void)fps_idx;
}
